// GNN_42417097015830
// MI455X (gfx1250) — matrix operand order repaired and checked by reading, not compiled, not run on hardware
//
#include <hip/hip_runtime.h>
#include <hip/hip_bf16.h>

// ---------------------------------------------------------------------------
// GENConv DeepGCN for MI455X (gfx1250, wave32).
// GEMMs: v_wmma_f32_16x16x32_bf16; B-matrix staged into LDS with
// global_load_async_to_lds_b128 (ASYNCcnt, double-buffered pipeline);
// activations converted f32->bf16 on the fly; f32 accumulate.
// Edge scatter-softmax: one wave per edge, float4 loads + f32 global atomics.
// LayerNorm: one wave per row, __shfl_xor wave32 reductions.
// ---------------------------------------------------------------------------

typedef __bf16 bf16;
typedef bf16  v16bf __attribute__((ext_vector_type(16)));
typedef bf16  v8bf  __attribute__((ext_vector_type(8)));
typedef float v8f   __attribute__((ext_vector_type(8)));

#define NN  40000
#define EE  600000
#define GG  64
#define HH  128
#define LL  3

#define SB_STRIDE 40   // bf16 per column slice (32 K + 8 pad): 80B, 16B-aligned

// ------------------------- WMMA fragment loaders ---------------------------
// A: 16x32 bf16 tile from f32 row-major [M,K].  Per ISA layout:
//   lane = 16*half + row ; K = (v>=4?16:0) + half*8 + 2*(v&3) + p
//   -> two contiguous 8-float chunks per lane.
__device__ __forceinline__ v16bf load_a_frag(const float* __restrict__ A,
                                             int lda, int m0, int k0, int lane) {
  const int row  = m0 + (lane & 15);
  const int half = lane >> 4;
  const float* p = A + (size_t)row * lda + k0 + half * 8;
  float4 c0 = *(const float4*)(p);
  float4 c1 = *(const float4*)(p + 4);
  float4 c2 = *(const float4*)(p + 16);
  float4 c3 = *(const float4*)(p + 20);
  v16bf a;
  a[0]  = (bf16)c0.x; a[1]  = (bf16)c0.y; a[2]  = (bf16)c0.z; a[3]  = (bf16)c0.w;
  a[4]  = (bf16)c1.x; a[5]  = (bf16)c1.y; a[6]  = (bf16)c1.z; a[7]  = (bf16)c1.w;
  a[8]  = (bf16)c2.x; a[9]  = (bf16)c2.y; a[10] = (bf16)c2.z; a[11] = (bf16)c2.w;
  a[12] = (bf16)c3.x; a[13] = (bf16)c3.y; a[14] = (bf16)c3.z; a[15] = (bf16)c3.w;
  return a;
}

// B fragment (32x16) read from the LDS-staged slice: lane (col = lane&15,
// Kbase = 8*(lane>>4)) reads K=[Kbase,Kbase+8) and [Kbase+16,Kbase+24) as two 16B halves.
__device__ __forceinline__ v16bf load_b_frag_lds(const bf16* sb, int ncol_local,
                                                 int lane) {
  const int col = ncol_local + (lane & 15);
  const int kb  = (lane >> 4) * 8;
  const bf16* p = sb + col * SB_STRIDE + kb;
  v8bf lo = *(const v8bf*)(p);
  v8bf hi = *(const v8bf*)(p + 16);
  return __builtin_shufflevector(lo, hi, 0, 1, 2, 3, 4, 5, 6, 7,
                                 8, 9, 10, 11, 12, 13, 14, 15);
}

// Async-stage one 128-col x 32-K bf16 slice of Wt[Nout,K] into LDS.
// 8KB = 512 x 16B chunks; 256 threads issue 2 async b128 copies each.
__device__ __forceinline__ void stage_b_async(const bf16* __restrict__ Wt, int K,
                                              int nblk, int k0, bf16* sbuf) {
  const unsigned lds_base = (unsigned)(size_t)sbuf;   // low 32b = LDS offset
  const int tid = threadIdx.x;
#pragma unroll
  for (int c = 0; c < 2; ++c) {
    const int chunk = tid + c * 256;          // 0..511
    const int col   = chunk >> 2;             // 4 chunks per 32-K column slice
    const int ko    = (chunk & 3) * 8;        // bf16 offset within slice
    const bf16* g   = Wt + (size_t)(nblk + col) * K + (k0 + ko);
    const unsigned l = lds_base + (unsigned)((col * SB_STRIDE + ko) * 2);
    asm volatile("global_load_async_to_lds_b128 %0, %1, off"
                 :: "v"(l), "v"(g) : "memory");
  }
}

// C = A[M,K](f32) @ W[K,Nout] + bias (+ optional addsrc residual).
// Block = 256 threads = 8 waves; block tile 64 rows x 128 cols, wave tile
// 16 rows x 64 cols (4 accumulators).  M%64==0, K%32==0, Nout%128==0.
extern "C" __global__ void __launch_bounds__(256)
gemm_bf16_wmma(const float* __restrict__ A, const bf16* __restrict__ Wt,
               const float* __restrict__ bias, const float* __restrict__ addsrc,
               float* __restrict__ C, int K, int Nout) {
  __shared__ __align__(32) bf16 sB[2][128 * SB_STRIDE];   // 2 x 10.0 KB

  const int lane = threadIdx.x & 31;
  const int wv   = threadIdx.x >> 5;
  const int m0   = blockIdx.x * 64 + (wv & 3) * 16;
  const int nblk = blockIdx.y * 128;
  const int nloc = (wv >> 2) * 64;          // wave's column offset in block
  const int n0   = nblk + nloc;

  v8f zero = {};
  v8f acc[4] = {zero, zero, zero, zero};

  stage_b_async(Wt, K, nblk, 0, sB[0]);
  const int KC = K >> 5;
  for (int kc = 0; kc < KC; ++kc) {
    asm volatile("s_wait_asynccnt 0x0" ::: "memory");   // own async copies done
    __syncthreads();                                    // everyone's copies done
    if (kc + 1 < KC)
      stage_b_async(Wt, K, nblk, (kc + 1) << 5, sB[(kc + 1) & 1]);

    const int k0 = kc << 5;
    if (k0 + 32 < K)   // pull the next A chunk toward the caches
      __builtin_prefetch(A + (size_t)(m0 + (lane & 15)) * K + k0 + 32, 0, 0);
    v16bf a = load_a_frag(A, K, m0, k0, lane);
    const bf16* sb = sB[kc & 1];
#pragma unroll
    for (int j = 0; j < 4; ++j) {
      v16bf b = load_b_frag_lds(sb, nloc + j * 16, lane);
      acc[j] = __builtin_amdgcn_wmma_f32_16x16x32_bf16(
          /*neg_a=*/false, a, /*neg_b=*/false, b,
          /*c_mod=*/(short)0, acc[j], /*reuse_a=*/false, /*reuse_b=*/false);
    }
    // top-of-loop wait+barrier of the NEXT iteration protects this buffer from
    // being overwritten before all waves finish reading it: no barrier here.
  }

  // D layout: VGPR r, lane l -> M = m0 + r + 8*(l>=16), N = n-tile + (l&15)
  const int half = lane >> 4, nl = lane & 15;
#pragma unroll
  for (int j = 0; j < 4; ++j) {
    const int n = n0 + j * 16 + nl;
    const float bv = bias[n];
#pragma unroll
    for (int r = 0; r < 8; ++r) {
      const int m = m0 + half * 8 + r;
      const size_t idx = (size_t)m * Nout + n;
      float v = acc[j][r] + bv;
      if (addsrc) v += addsrc[idx];
      C[idx] = v;
    }
  }
}

// Transpose + convert weights: W[K,Nout] f32 -> Wt[Nout,K] bf16.
extern "C" __global__ void wcvt_kernel(const float* __restrict__ W,
                                       bf16* __restrict__ Wt, int K, int Nout) {
  int idx = blockIdx.x * 256 + threadIdx.x;
  if (idx < K * Nout) {
    int k = idx / Nout, n = idx % Nout;
    Wt[(size_t)n * K + k] = (bf16)W[idx];
  }
}

// ------------------------- scatter-softmax passes --------------------------
__device__ __forceinline__ void atomicMaxF32(float* addr, float val) {
  int cur = __float_as_int(*addr);
  while (__int_as_float(cur) < val) {
    int prev = atomicCAS((int*)addr, cur, __float_as_int(val));
    if (prev == cur) break;
    cur = prev;
  }
}

extern "C" __global__ void seg_init_kernel(float* __restrict__ mmax,
                                           float* __restrict__ den,
                                           float* __restrict__ num) {
  int i = blockIdx.x * 256 + threadIdx.x;
  if (i < NN * HH) { mmax[i] = -3.0e38f; den[i] = 0.f; num[i] = 0.f; }
}

// One wave per edge; lane owns 4 contiguous features (float4).
extern "C" __global__ void __launch_bounds__(256)
edge_max_kernel(const float* __restrict__ xn, const float* __restrict__ eattr,
                const float* __restrict__ ew, const float* __restrict__ eb,
                const int* __restrict__ src, const int* __restrict__ dst,
                const float* __restrict__ t, int layer, float* __restrict__ mmax) {
  int e = blockIdx.x * 8 + (threadIdx.x >> 5);
  if (e >= EE) return;
  const int lane = threadIdx.x & 31;
  const int s = src[e], d = dst[e];
  const float ea = eattr[e], tt = t[layer];
  float4 x = ((const float4*)(xn + (size_t)s * HH))[lane];
  float4 w = ((const float4*)ew)[lane];
  float4 b = ((const float4*)eb)[lane];
  float* mp = mmax + (size_t)d * HH + lane * 4;
  atomicMaxF32(mp + 0, (fmaxf(x.x + ea * w.x + b.x, 0.f) + 1e-7f) * tt);
  atomicMaxF32(mp + 1, (fmaxf(x.y + ea * w.y + b.y, 0.f) + 1e-7f) * tt);
  atomicMaxF32(mp + 2, (fmaxf(x.z + ea * w.z + b.z, 0.f) + 1e-7f) * tt);
  atomicMaxF32(mp + 3, (fmaxf(x.w + ea * w.w + b.w, 0.f) + 1e-7f) * tt);
}

extern "C" __global__ void __launch_bounds__(256)
edge_sum_kernel(const float* __restrict__ xn, const float* __restrict__ eattr,
                const float* __restrict__ ew, const float* __restrict__ eb,
                const int* __restrict__ src, const int* __restrict__ dst,
                const float* __restrict__ t, int layer,
                const float* __restrict__ mmax,
                float* __restrict__ den, float* __restrict__ num) {
  int e = blockIdx.x * 8 + (threadIdx.x >> 5);
  if (e >= EE) return;
  const int lane = threadIdx.x & 31;
  const int s = src[e], d = dst[e];
  const float ea = eattr[e], tt = t[layer];
  float4 x  = ((const float4*)(xn + (size_t)s * HH))[lane];
  float4 w  = ((const float4*)ew)[lane];
  float4 b  = ((const float4*)eb)[lane];
  float4 mr = ((const float4*)(mmax + (size_t)d * HH))[lane];
  float* dp = den + (size_t)d * HH + lane * 4;
  float* np = num + (size_t)d * HH + lane * 4;
  float m0 = fmaxf(x.x + ea * w.x + b.x, 0.f) + 1e-7f;
  float m1 = fmaxf(x.y + ea * w.y + b.y, 0.f) + 1e-7f;
  float m2 = fmaxf(x.z + ea * w.z + b.z, 0.f) + 1e-7f;
  float m3 = fmaxf(x.w + ea * w.w + b.w, 0.f) + 1e-7f;
  float e0 = expf(m0 * tt - mr.x), e1 = expf(m1 * tt - mr.y);
  float e2 = expf(m2 * tt - mr.z), e3 = expf(m3 * tt - mr.w);
  atomicAdd(dp + 0, e0); atomicAdd(dp + 1, e1);
  atomicAdd(dp + 2, e2); atomicAdd(dp + 3, e3);
  atomicAdd(np + 0, e0 * m0); atomicAdd(np + 1, e1 * m1);
  atomicAdd(np + 2, e2 * m2); atomicAdd(np + 3, e3 * m3);
}

// o = num / max(den,1e-16) + xn   (root add). In-place safe (o may == xn).
extern "C" __global__ void combine_kernel(const float* __restrict__ num,
                                          const float* __restrict__ den,
                                          const float* __restrict__ xn,
                                          float* __restrict__ o) {
  int i = blockIdx.x * 256 + threadIdx.x;
  if (i < NN * HH) o[i] = num[i] / fmaxf(den[i], 1e-16f) + xn[i];
}

// ------------------------- LayerNorm + ReLU (wave per row) -----------------
extern "C" __global__ void __launch_bounds__(256)
ln_relu_kernel(const float* __restrict__ in, const float* __restrict__ g,
               const float* __restrict__ b, float* __restrict__ out, int W) {
  int row = blockIdx.x * 8 + (threadIdx.x >> 5);
  if (row >= NN) return;
  const int lane = threadIdx.x & 31;
  const float* rp = in + (size_t)row * W;
  float4 v0 = ((const float4*)rp)[lane];
  float4 v1 = {0.f, 0.f, 0.f, 0.f};
  float sum = v0.x + v0.y + v0.z + v0.w;
  float sq  = v0.x * v0.x + v0.y * v0.y + v0.z * v0.z + v0.w * v0.w;
  if (W == 256) {
    v1 = ((const float4*)rp)[lane + 32];
    sum += v1.x + v1.y + v1.z + v1.w;
    sq  += v1.x * v1.x + v1.y * v1.y + v1.z * v1.z + v1.w * v1.w;
  }
#pragma unroll
  for (int off = 16; off > 0; off >>= 1) {
    sum += __shfl_xor(sum, off, 32);
    sq  += __shfl_xor(sq,  off, 32);
  }
  const float mu  = sum / W;
  const float inv = rsqrtf(sq / W - mu * mu + 1e-5f);
  float* op = out + (size_t)row * W;
  int base = lane * 4;
  op[base + 0] = fmaxf((v0.x - mu) * inv * g[base + 0] + b[base + 0], 0.f);
  op[base + 1] = fmaxf((v0.y - mu) * inv * g[base + 1] + b[base + 1], 0.f);
  op[base + 2] = fmaxf((v0.z - mu) * inv * g[base + 2] + b[base + 2], 0.f);
  op[base + 3] = fmaxf((v0.w - mu) * inv * g[base + 3] + b[base + 3], 0.f);
  if (W == 256) {
    base = (lane + 32) * 4;
    op[base + 0] = fmaxf((v1.x - mu) * inv * g[base + 0] + b[base + 0], 0.f);
    op[base + 1] = fmaxf((v1.y - mu) * inv * g[base + 1] + b[base + 1], 0.f);
    op[base + 2] = fmaxf((v1.z - mu) * inv * g[base + 2] + b[base + 2], 0.f);
    op[base + 3] = fmaxf((v1.w - mu) * inv * g[base + 3] + b[base + 3], 0.f);
  }
}

// ------------------------- pooling + MLP head ------------------------------
extern "C" __global__ void pool_init_kernel(float* __restrict__ pooled,
                                            float* __restrict__ cnt) {
  int i = blockIdx.x * 256 + threadIdx.x;
  if (i < GG * HH) pooled[i] = 0.f;
  if (i < GG) cnt[i] = 0.f;
}

extern "C" __global__ void __launch_bounds__(256)
pool_accum_kernel(const float* __restrict__ hin, const int* __restrict__ batch,
                  float* __restrict__ pooled, float* __restrict__ cnt) {
  int n = blockIdx.x * 8 + (threadIdx.x >> 5);
  if (n >= NN) return;
  const int lane = threadIdx.x & 31;
  const int g = batch[n];
  float4 v = ((const float4*)(hin + (size_t)n * HH))[lane];
  float* pp = pooled + (size_t)g * HH + lane * 4;
  atomicAdd(pp + 0, v.x); atomicAdd(pp + 1, v.y);
  atomicAdd(pp + 2, v.z); atomicAdd(pp + 3, v.w);
  if (lane == 0) atomicAdd(&cnt[g], 1.0f);
}

extern "C" __global__ void __launch_bounds__(256)
head_kernel(const float* __restrict__ pooled, const float* __restrict__ cnt,
            const float* __restrict__ hw1, const float* __restrict__ hb1,
            const float* __restrict__ hw2, const float* __restrict__ hb2,
            const float* __restrict__ hw3, const float* __restrict__ hb3,
            float* __restrict__ out) {
  __shared__ float P[GG * HH];    // 32 KB
  __shared__ float T1[GG * 64];   // 16 KB
  __shared__ float T2[GG * 32];   //  8 KB
  const int tid = threadIdx.x;
  for (int i = tid; i < GG * HH; i += 256)
    P[i] = pooled[i] / fmaxf(cnt[i >> 7], 1.0f);
  __syncthreads();
  for (int i = tid; i < GG * 64; i += 256) {
    int g = i >> 6, o = i & 63;
    float s = hb1[o];
    for (int k = 0; k < 128; ++k) s += P[g * 128 + k] * hw1[k * 64 + o];
    T1[i] = fmaxf(s, 0.f);
  }
  __syncthreads();
  for (int i = tid; i < GG * 32; i += 256) {
    int g = i >> 5, o = i & 31;
    float s = hb2[o];
    for (int k = 0; k < 64; ++k) s += T1[g * 64 + k] * hw2[k * 32 + o];
    T2[i] = fmaxf(s, 0.f);
  }
  __syncthreads();
  for (int i = tid; i < GG; i += 256) {
    float s = hb3[0];
    for (int k = 0; k < 32; ++k) s += T2[i * 32 + k] * hw3[k];
    out[i] = s;
  }
}

// --------------------------------- driver ----------------------------------
extern "C" void kernel_launch(void* const* d_in, const int* in_sizes, int n_in,
                              void* d_out, int out_size, void* d_ws, size_t ws_size,
                              hipStream_t stream) {
  (void)in_sizes; (void)n_in; (void)out_size; (void)ws_size;
  const float* x         = (const float*)d_in[0];
  const float* edge_attr = (const float*)d_in[1];
  const float* node_w    = (const float*)d_in[2];
  const float* node_b    = (const float*)d_in[3];
  const float* edge_w    = (const float*)d_in[4];
  const float* edge_b    = (const float*)d_in[5];
  const float* t         = (const float*)d_in[6];
  const float* lin1_w    = (const float*)d_in[7];
  const float* lin1_b    = (const float*)d_in[8];
  const float* ln1_g     = (const float*)d_in[9];
  const float* ln1_bt    = (const float*)d_in[10];
  const float* lin2_w    = (const float*)d_in[11];
  const float* lin2_b    = (const float*)d_in[12];
  const float* blk_g     = (const float*)d_in[13];
  const float* blk_b     = (const float*)d_in[14];
  const float* hw1       = (const float*)d_in[15];
  const float* hb1       = (const float*)d_in[16];
  const float* hw2       = (const float*)d_in[17];
  const float* hb2       = (const float*)d_in[18];
  const float* hw3       = (const float*)d_in[19];
  const float* hb3       = (const float*)d_in[20];
  const int* edge_index  = (const int*)d_in[21];
  const int* batch       = (const int*)d_in[22];
  const int* srcI = edge_index;
  const int* dstI = edge_index + EE;

  // ---- workspace layout (mmax/den/num alias tmp256: disjoint live ranges) --
  char* ws = (char*)d_ws;
  size_t off = 0;
  auto alloc = [&](size_t bytes) -> void* {
    void* p = ws + off;
    off += (bytes + 255) & ~(size_t)255;
    return p;
  };
  const size_t NH4 = (size_t)NN * HH * 4;
  float* h      = (float*)alloc(NH4);
  float* rbuf   = (float*)alloc(NH4);
  char*  uni    = (char*)alloc(3 * NH4);          // 61.4 MB union region
  float* mmax   = (float*)uni;
  float* den    = (float*)(uni + NH4);
  float* num    = (float*)(uni + 2 * NH4);
  float* tmp256 = (float*)uni;                    // 41 MB, reuses union
  float* pooled = (float*)alloc((size_t)GG * HH * 4);
  float* cnt    = (float*)alloc((size_t)GG * 4);
  bf16* node_wT = (bf16*)alloc((size_t)128 * 128 * 2);
  bf16* lin1_wT = (bf16*)alloc((size_t)3 * 256 * 128 * 2);
  bf16* lin2_wT = (bf16*)alloc((size_t)3 * 128 * 256 * 2);

  const dim3 b256(256);
  const int NHB = (NN * HH) / 256;   // 20000 blocks
  const int EBL = EE / 8;            // 75000 blocks (8 edges per block)

  // ---- weight convert + transpose to bf16 ----------------------------------
  wcvt_kernel<<<(128 * 128 + 255) / 256, b256, 0, stream>>>(node_w, node_wT, 128, 128);
  for (int i = 0; i < LL; ++i) {
    wcvt_kernel<<<(128 * 256 + 255) / 256, b256, 0, stream>>>(
        lin1_w + (size_t)i * 128 * 256, lin1_wT + (size_t)i * 256 * 128, 128, 256);
    wcvt_kernel<<<(256 * 128 + 255) / 256, b256, 0, stream>>>(
        lin2_w + (size_t)i * 256 * 128, lin2_wT + (size_t)i * 128 * 256, 256, 128);
  }

  // ---- node encoder: h = x @ node_w + node_b -------------------------------
  gemm_bf16_wmma<<<dim3(NN / 64, 1), b256, 0, stream>>>(
      x, node_wT, node_b, nullptr, h, 128, 128);

  // ---- GENConv layers ------------------------------------------------------
  for (int i = 0; i < LL; ++i) {
    const float* xin;
    if (i == 0) {
      xin = h;                                    // bare conv
    } else {
      // res+ pre-step: r = relu(LN(h))
      ln_relu_kernel<<<NN / 8, b256, 0, stream>>>(h, blk_g + i * HH, blk_b + i * HH, rbuf, HH);
      xin = rbuf;
    }
    // scatter-softmax aggregation
    seg_init_kernel<<<NHB, b256, 0, stream>>>(mmax, den, num);
    edge_max_kernel<<<EBL, b256, 0, stream>>>(xin, edge_attr, edge_w, edge_b,
                                              srcI, dstI, t, i, mmax);
    edge_sum_kernel<<<EBL, b256, 0, stream>>>(xin, edge_attr, edge_w, edge_b,
                                              srcI, dstI, t, i, mmax, den, num);
    combine_kernel<<<NHB, b256, 0, stream>>>(num, den, xin, rbuf);
    // MLP: lin1 -> LN -> ReLU -> lin2 (+residual for layers >= 1)
    gemm_bf16_wmma<<<dim3(NN / 64, 2), b256, 0, stream>>>(
        rbuf, lin1_wT + (size_t)i * 256 * 128, lin1_b + i * 256, nullptr, tmp256, 128, 256);
    ln_relu_kernel<<<NN / 8, b256, 0, stream>>>(tmp256, ln1_g + i * 256, ln1_bt + i * 256, tmp256, 256);
    gemm_bf16_wmma<<<dim3(NN / 64, 1), b256, 0, stream>>>(
        tmp256, lin2_wT + (size_t)i * 128 * 256, lin2_b + i * HH,
        (i == 0) ? nullptr : h, h, 256, 128);
  }

  // ---- final norm/act (layer 0 params) + mean pool + head ------------------
  ln_relu_kernel<<<NN / 8, b256, 0, stream>>>(h, blk_g, blk_b, rbuf, HH);
  pool_init_kernel<<<(GG * HH + 255) / 256, b256, 0, stream>>>(pooled, cnt);
  pool_accum_kernel<<<NN / 8, b256, 0, stream>>>(rbuf, batch, pooled, cnt);
  head_kernel<<<1, b256, 0, stream>>>(pooled, cnt, hw1, hb1, hw2, hb2, hw3, hb3,
                                      (float*)d_out);
}
